// TrackFormerDecoderLayer_47777216201312
// MI455X (gfx1250) — matrix operand order repaired and checked by reading, not compiled, not run on hardware
//
#include <hip/hip_runtime.h>
#include <hip/hip_bf16.h>

typedef __attribute__((ext_vector_type(16))) __bf16 v16bf;
typedef __attribute__((ext_vector_type(8)))  float  v8f;
typedef int v4i_gcc __attribute__((vector_size(16)));   // matches builtin's V4i

#define AS_GLOBAL __attribute__((address_space(1)))
#define AS_LOCAL  __attribute__((address_space(3)))

#if defined(__has_builtin)
#if __has_builtin(__builtin_amdgcn_global_load_async_to_lds_b128)
#define HAVE_ASYNC_LDS 1
#endif
#endif

// ---------- helpers ----------

__device__ __forceinline__ unsigned short f2bf(float f) {
  union { float f; unsigned u; } c; c.f = f;
  unsigned u = c.u;
  u += 0x7FFFu + ((u >> 16) & 1u);   // round-to-nearest-even
  return (unsigned short)(u >> 16);
}

__device__ __forceinline__ float bf2f(unsigned short h) {
  union { unsigned u; float f; } c; c.u = ((unsigned)h) << 16;
  return c.f;
}

// 16-byte global -> LDS copy; async (ASYNCcnt) when the builtin exists.
__device__ __forceinline__ void cp16(const unsigned short* __restrict__ g,
                                     unsigned short* __restrict__ l) {
#ifdef HAVE_ASYNC_LDS
  __builtin_amdgcn_global_load_async_to_lds_b128(
      (AS_GLOBAL v4i_gcc*)(g), (AS_LOCAL v4i_gcc*)(l), 0, 0);
#else
  *(uint4*)l = *(const uint4*)g;
#endif
}

__device__ __forceinline__ void zero16(unsigned short* l) {
  uint4 z = {0u, 0u, 0u, 0u};
  *(uint4*)l = z;
}

__device__ __forceinline__ void async_join() {
#ifdef HAVE_ASYNC_LDS
#if __has_builtin(__builtin_amdgcn_s_wait_asynccnt)
  __builtin_amdgcn_s_wait_asynccnt(0);
#else
  asm volatile("s_wait_asynccnt 0x0" ::: "memory");
#endif
#endif
}

// A fragment (16x32 bf16): lane<16 -> row=lane, K {0..7,16..23};
// lane>=16 -> row=lane-16, K {8..15,24..31}.  smem row-major ldk=32.
__device__ __forceinline__ v16bf load_frag_a(const unsigned short* smem, int row, int lane) {
  const unsigned short* p = smem + row * 32 + ((lane >> 4) & 1) * 8;
  union { uint4 u[2]; v16bf v; } f;
  f.u[0] = *(const uint4*)(p);
  f.u[1] = *(const uint4*)(p + 16);
  return f.v;
}

// B fragment (32x16 bf16), smem holds B^T ([col][k], ldk=32):
// lane<16 -> col=lane, K {0..7,16..23} ; lane>=16 -> col=lane-16, K {8..15,24..31}.
__device__ __forceinline__ v16bf load_frag_b(const unsigned short* smem, int col, int lane) {
  const unsigned short* p = smem + col * 32 + ((lane >> 4) & 1) * 8;
  union { uint4 u[2]; v16bf v; } f;
  f.u[0] = *(const uint4*)(p);
  f.u[1] = *(const uint4*)(p + 16);
  return f.v;
}

__device__ __forceinline__ v8f wmma_bf16(v16bf a, v16bf b, v8f c) {
  return __builtin_amdgcn_wmma_f32_16x16x32_bf16(false, a, false, b, (short)0, c, false, false);
}

// ---------- WMMA GEMM:  C = A(MxK) @ B(KxN) + bias ----------
// A: bf16 row-major.  Bt: bf16 N x K row-major (B transposed).
// EPI: 0 = bias, 1 = bias+relu, 2 = bias then * rowmask[row]
template <int EPI>
__global__ __launch_bounds__(128)
void gemm_bf16(const unsigned short* __restrict__ A,
               const unsigned short* __restrict__ Bt,
               const float* __restrict__ bias, const float* __restrict__ rowmask,
               float* __restrict__ C, unsigned short* __restrict__ Cbf,
               int M, int N, int K)
{
  constexpr int BM = 64, BN = 64, BK = 32;
  __shared__ alignas(16) unsigned short sA[BM * BK];     // [row][k]
  __shared__ alignas(16) unsigned short sB[BN * BK];     // [col][k]
  const int tid  = threadIdx.x;
  const int lane = tid & 31;
  const int wave = tid >> 5;
  const int wm = (wave >> 1) * 32;
  const int wn = (wave & 1) * 32;
  const int m0 = blockIdx.y * BM;
  const int n0 = blockIdx.x * BN;
  const int lr = lane & 15;
  v8f acc[2][2] = {};

  for (int k0 = 0; k0 < K; k0 += BK) {
    // stage A tile: 64 rows x 32 k  = 256 x 16B chunks
    for (int c = 0; c < 2; ++c) {
      int cc = tid + c * 128;
      int r  = cc >> 2;
      int off = (cc & 3) << 3;               // element offset (8 bf16 = 16B)
      int gm = m0 + r;
      if (gm < M) cp16(A + (size_t)gm * K + k0 + off, sA + r * BK + off);
      else        zero16(sA + r * BK + off);
    }
    // stage B^T tile: 64 cols x 32 k
    for (int c = 0; c < 2; ++c) {
      int cc = tid + c * 128;
      int col = cc >> 2;
      int off = (cc & 3) << 3;
      int gn = n0 + col;
      if (gn < N) cp16(Bt + (size_t)gn * K + k0 + off, sB + col * BK + off);
      else        zero16(sB + col * BK + off);
    }
    if (k0 + BK < K) {                        // global_prefetch hint, next A tile
      int pr = m0 + (tid >> 1);
      if (pr < M) __builtin_prefetch(A + (size_t)pr * K + k0 + BK, 0, 0);
    }
    async_join();
    __syncthreads();

    v16bf af[2], bf[2];
    for (int i = 0; i < 2; ++i) af[i] = load_frag_a(sA, wm + i * 16 + lr, lane);
    for (int j = 0; j < 2; ++j) bf[j] = load_frag_b(sB, wn + j * 16 + lr, lane);
    for (int i = 0; i < 2; ++i)
      for (int j = 0; j < 2; ++j)
        acc[i][j] = wmma_bf16(af[i], bf[j], acc[i][j]);
    __syncthreads();
  }

  const int half = (lane >> 4) & 1;
  for (int i = 0; i < 2; ++i)
    for (int j = 0; j < 2; ++j) {
      int col = n0 + wn + j * 16 + lr;
      if (col >= N) continue;
      float bvv = bias ? bias[col] : 0.f;
      for (int r = 0; r < 8; ++r) {
        int row = m0 + wm + i * 16 + half * 8 + r;
        if (row >= M) continue;
        float x = acc[i][j][r] + bvv;
        if (EPI == 1) x = fmaxf(x, 0.f);
        if (EPI == 2) x *= rowmask[row];
        size_t o = (size_t)row * N + col;
        if (C)   C[o]   = x;
        if (Cbf) Cbf[o] = f2bf(x);
      }
    }
}

// ---------- flash MHA on bf16 Q/K/V (HD=32, Q=K=900) ----------
__global__ __launch_bounds__(32)
void mha_flash(const unsigned short* __restrict__ qh, const unsigned short* __restrict__ kh,
               const unsigned short* __restrict__ vh, const float* __restrict__ mask,
               unsigned short* __restrict__ out)
{
  const int qt = blockIdx.x;          // 0..56
  const int h  = blockIdx.y;          // 0..7
  const int n  = blockIdx.z;          // 0..3
  const int lane = threadIdx.x;
  const int lr = lane & 15, half = (lane >> 4) & 1;
  __shared__ alignas(16) unsigned short sQ[16 * 32];
  __shared__ alignas(16) unsigned short sK[32 * 32];   // [key][feat]
  __shared__ alignas(16) unsigned short sV[32 * 32];   // [feat][key]
  __shared__ alignas(16) unsigned short sP[16 * 32];
  const int q0 = qt * 16;
  const float qscale = 0.17677669529663687f;           // 1/sqrt(32)

  for (int cc = lane; cc < 64; cc += 32) {             // Q tile: 16 x 32 bf16
    int r = cc >> 2, off = (cc & 3) << 3;
    int q = q0 + r;
    if (q < 900) cp16(qh + ((size_t)(n * 900 + q)) * 256 + h * 32 + off, sQ + r * 32 + off);
    else         zero16(sQ + r * 32 + off);
  }
  async_join();
  __syncthreads();
  v16bf qf = load_frag_a(sQ, lr, lane);

  v8f o0 = {}, o1 = {};
  float mrow[8], lrow[8];
  for (int r = 0; r < 8; ++r) { mrow[r] = -1e30f; lrow[r] = 0.f; }

  for (int kt = 0; kt < 29; ++kt) {
    int kt0 = kt * 32;
    for (int cc = lane; cc < 128; cc += 32) {          // K tile copy
      int kk = cc >> 2, off = (cc & 3) << 3;
      int key = kt0 + kk;
      if (key < 900) cp16(kh + ((size_t)(n * 900 + key)) * 256 + h * 32 + off, sK + kk * 32 + off);
      else           zero16(sK + kk * 32 + off);
    }
    for (int idx = lane; idx < 1024; idx += 32) {      // V tile, transposed
      int kk = idx >> 5, cf = idx & 31;
      int key = kt0 + kk;
      unsigned short v = 0;
      if (key < 900) v = vh[((size_t)(n * 900 + key)) * 256 + h * 32 + cf];
      sV[cf * 32 + kk] = v;
    }
    async_join();
    __syncthreads();

    v16bf kf0 = load_frag_b(sK, 0 + lr, lane);
    v16bf kf1 = load_frag_b(sK, 16 + lr, lane);
    v8f zc = {};
    v8f s0 = wmma_bf16(qf, kf0, zc);
    v8f s1 = wmma_bf16(qf, kf1, zc);

    int key0 = kt0 + lr, key1 = kt0 + 16 + lr;
    bool ok0 = (key0 < 900) && (mask[n * 900 + key0] > 0.f);
    bool ok1 = (key1 < 900) && (mask[n * 900 + key1] > 0.f);
    float p0[8], p1[8];
    for (int r = 0; r < 8; ++r) {
      float a = ok0 ? s0[r] * qscale : -1e30f;
      float b = ok1 ? s1[r] * qscale : -1e30f;
      float lm = fmaxf(a, b);
      for (int d = 1; d < 16; d <<= 1) lm = fmaxf(lm, __shfl_xor(lm, d, 32));
      float mnew = fmaxf(mrow[r], lm);
      float sc = __expf(mrow[r] - mnew);
      mrow[r] = mnew;
      float e0 = __expf(a - mnew), e1 = __expf(b - mnew);
      p0[r] = e0; p1[r] = e1;
      float ls = e0 + e1;
      for (int d = 1; d < 16; d <<= 1) ls += __shfl_xor(ls, d, 32);
      lrow[r] = lrow[r] * sc + ls;
      o0[r] *= sc; o1[r] *= sc;
    }
    for (int r = 0; r < 8; ++r) {
      int row = half * 8 + r;
      sP[row * 32 + lr]      = f2bf(p0[r]);
      sP[row * 32 + 16 + lr] = f2bf(p1[r]);
    }
    __syncthreads();
    v16bf pf  = load_frag_a(sP, lr, lane);
    v16bf vf0 = load_frag_b(sV, 0 + lr, lane);
    v16bf vf1 = load_frag_b(sV, 16 + lr, lane);
    o0 = wmma_bf16(pf, vf0, o0);
    o1 = wmma_bf16(pf, vf1, o1);
    __syncthreads();
  }

  for (int r = 0; r < 8; ++r) {
    int q = q0 + half * 8 + r;
    if (q < 900) {
      float inv = 1.0f / lrow[r];
      size_t base = ((size_t)(n * 900 + q)) * 256 + h * 32;
      out[base + lr]      = f2bf(o0[r] * inv);
      out[base + 16 + lr] = f2bf(o1[r] * inv);
    }
  }
}

// ---------- deformable bilinear sampling (bf16 value, P=4, Hh=8, HD=32) ----------
__global__ __launch_bounds__(256)
void deform_sample(const unsigned short* __restrict__ value, const float* __restrict__ offb,
                   const float* __restrict__ awb, const float* __restrict__ ref,
                   unsigned short* __restrict__ out)
{
  int row = blockIdx.x;               // n*900+q
  int n = row / 900;
  int t = threadIdx.x;
  int h = t >> 5, d = t & 31;

  float a0 = awb[row * 32 + h * 4 + 0];
  float a1 = awb[row * 32 + h * 4 + 1];
  float a2 = awb[row * 32 + h * 4 + 2];
  float a3 = awb[row * 32 + h * 4 + 3];
  float mx = fmaxf(fmaxf(a0, a1), fmaxf(a2, a3));
  float e0 = __expf(a0 - mx), e1 = __expf(a1 - mx), e2 = __expf(a2 - mx), e3 = __expf(a3 - mx);
  float inv = 1.f / (e0 + e1 + e2 + e3);
  float aw[4] = {e0 * inv, e1 * inv, e2 * inv, e3 * inv};

  float rx = ref[row * 2 + 0], ry = ref[row * 2 + 1];
  float accd = 0.f;
  for (int p = 0; p < 4; ++p) {
    float ox = offb[row * 64 + (h * 4 + p) * 2 + 0];
    float oy = offb[row * 64 + (h * 4 + p) * 2 + 1];
    float x = (rx + ox * (1.f / 200.f)) * 200.f - 0.5f;
    float y = (ry + oy * (1.f / 200.f)) * 200.f - 0.5f;
    float x0 = floorf(x), y0 = floorf(y);
    float g = 0.f;
    for (int dy = 0; dy < 2; ++dy)
      for (int dx = 0; dx < 2; ++dx) {
        float xc = x0 + dx, yc = y0 + dy;
        if (xc >= 0.f && xc < 200.f && yc >= 0.f && yc < 200.f) {
          float wgt = (1.f - fabsf(x - xc)) * (1.f - fabsf(y - yc));
          int idx = (int)yc * 200 + (int)xc;
          g += wgt * bf2f(value[((size_t)(n * 40000 + idx)) * 256 + h * 32 + d]);
        }
      }
    accd += aw[p] * g;
  }
  out[(size_t)row * 256 + t] = f2bf(accd);
}

// ---------- fused residual + LayerNorm, D=256 ----------
__global__ __launch_bounds__(256)
void add_ln(const float* __restrict__ a, const float* __restrict__ b,
            const float* __restrict__ g, const float* __restrict__ beta,
            float* __restrict__ out, unsigned short* __restrict__ obf)
{
  int row = blockIdx.x;
  int t = threadIdx.x;
  float x = a[(size_t)row * 256 + t] + b[(size_t)row * 256 + t];
  float s = x, ss = x * x;
  for (int d = 1; d < 32; d <<= 1) { s += __shfl_xor(s, d, 32); ss += __shfl_xor(ss, d, 32); }
  __shared__ float red[16];
  int wv = t >> 5;
  if ((t & 31) == 0) { red[wv] = s; red[8 + wv] = ss; }
  __syncthreads();
  float tot = 0.f, tot2 = 0.f;
  for (int i = 0; i < 8; ++i) { tot += red[i]; tot2 += red[8 + i]; }
  float mean = tot * (1.f / 256.f);
  float var  = tot2 * (1.f / 256.f) - mean * mean;
  float r    = rsqrtf(var + 1e-5f);
  float y = (x - mean) * r * g[t] + beta[t];
  out[(size_t)row * 256 + t] = y;
  if (obf) obf[(size_t)row * 256 + t] = f2bf(y);
}

__global__ __launch_bounds__(256)
void add2_bf(const float* __restrict__ a, const float* __restrict__ b,
             unsigned short* __restrict__ o, int n)
{
  int i = blockIdx.x * 256 + threadIdx.x;
  if (i < n) o[i] = f2bf(a[i] + b[i]);
}

__global__ __launch_bounds__(256)
void cvt_bf(const float* __restrict__ a, unsigned short* __restrict__ o, int n)
{
  int i = blockIdx.x * 256 + threadIdx.x;
  if (i < n) o[i] = f2bf(a[i]);
}

// Wt[n*K+k] = bf16(W[k*N+n])  — transpose-convert a KxN weight to NxK bf16
__global__ __launch_bounds__(256)
void cvt_bf_T(const float* __restrict__ W, unsigned short* __restrict__ Wt, int K, int N)
{
  int i = blockIdx.x * 256 + threadIdx.x;
  if (i < N * K) {
    int ncol = i / K, k = i - ncol * K;
    Wt[i] = f2bf(W[(size_t)k * N + ncol]);
  }
}

// ---------- launcher ----------
extern "C" void kernel_launch(void* const* d_in, const int* in_sizes, int n_in,
                              void* d_out, int out_size, void* d_ws, size_t ws_size,
                              hipStream_t stream) {
  (void)in_sizes; (void)n_in; (void)out_size; (void)ws_size;
  const float* queries = (const float*)d_in[0];
  const float* bev     = (const float*)d_in[1];
  const float* ref     = (const float*)d_in[2];
  const float* og      = (const float*)d_in[3];
  const float* mask    = (const float*)d_in[4];
  const float* Wq = (const float*)d_in[5];   const float* bq = (const float*)d_in[6];
  const float* Wk = (const float*)d_in[7];   const float* bk = (const float*)d_in[8];
  const float* Wv = (const float*)d_in[9];   const float* bvp = (const float*)d_in[10];
  const float* Wo = (const float*)d_in[11];  const float* bo = (const float*)d_in[12];
  const float* ln1g = (const float*)d_in[13]; const float* ln1b = (const float*)d_in[14];
  const float* Wval = (const float*)d_in[15]; const float* bval = (const float*)d_in[16];
  const float* Woff = (const float*)d_in[17]; const float* boff = (const float*)d_in[18];
  const float* Wattn = (const float*)d_in[19]; const float* battn = (const float*)d_in[20];
  const float* Wdo = (const float*)d_in[21]; const float* bdo = (const float*)d_in[22];
  const float* ln2g = (const float*)d_in[23]; const float* ln2b = (const float*)d_in[24];
  const float* W1 = (const float*)d_in[25];  const float* b1 = (const float*)d_in[26];
  const float* W2 = (const float*)d_in[27];  const float* b2 = (const float*)d_in[28];
  const float* ln3g = (const float*)d_in[29]; const float* ln3b = (const float*)d_in[30];
  float* out = (float*)d_out;

  const size_t S1 = (size_t)3600 * 256;      // 921600
  float* w = (float*)d_ws;
  float* out1 = w + 0 * S1;
  float* out2 = w + 1 * S1;
  float* out3 = w + 2 * S1;
  float* out4 = w + 3 * S1;
  float* out5 = w + 4 * S1;
  float* offb = w + 5 * S1;                  // 3600*64
  float* awb  = offb + (size_t)3600 * 64;    // 3600*32
  unsigned short* us = (unsigned short*)(awb + (size_t)3600 * 32);
  unsigned short* qbf    = us + 0 * S1;
  unsigned short* qkbf   = us + 1 * S1;
  unsigned short* qhbf   = us + 2 * S1;
  unsigned short* khbf   = us + 3 * S1;
  unsigned short* vhbf   = us + 4 * S1;
  unsigned short* attnbf = us + 5 * S1;
  unsigned short* augbf  = us + 6 * S1;
  unsigned short* sampbf = us + 7 * S1;
  unsigned short* o4bf   = us + 8 * S1;
  unsigned short* ffn1bf = us + 9 * S1;                      // 3600*512
  unsigned short* bevbf  = ffn1bf + (size_t)3600 * 512;      // 160000*256
  unsigned short* valbf  = bevbf  + (size_t)160000 * 256;    // 160000*256
  unsigned short* wqt  = valbf + (size_t)160000 * 256;       // 256*256 each below
  unsigned short* wkt  = wqt  + 65536;
  unsigned short* wvt  = wkt  + 65536;
  unsigned short* wot  = wvt  + 65536;
  unsigned short* wvalt = wot + 65536;
  unsigned short* wdot = wvalt + 65536;
  unsigned short* wofft = wdot + 65536;                      // 64*256
  unsigned short* wattnt = wofft + 16384;                    // 32*256
  unsigned short* w1t  = wattnt + 8192;                      // 512*256
  unsigned short* w2t  = w1t + 131072;                       // 256*512

  dim3 b128(128), b256(256), b32(32);
  auto gg = [](int M, int N) { return dim3((unsigned)((N + 63) / 64), (unsigned)((M + 63) / 64)); };
  auto ge = [](int n) { return dim3((unsigned)((n + 255) / 256)); };

  // one-time operand conversions (bf16), weight transposes
  cvt_bf_T<<<ge(256 * 256), b256, 0, stream>>>(Wq,   wqt,   256, 256);
  cvt_bf_T<<<ge(256 * 256), b256, 0, stream>>>(Wk,   wkt,   256, 256);
  cvt_bf_T<<<ge(256 * 256), b256, 0, stream>>>(Wv,   wvt,   256, 256);
  cvt_bf_T<<<ge(256 * 256), b256, 0, stream>>>(Wo,   wot,   256, 256);
  cvt_bf_T<<<ge(256 * 256), b256, 0, stream>>>(Wval, wvalt, 256, 256);
  cvt_bf_T<<<ge(256 * 256), b256, 0, stream>>>(Wdo,  wdot,  256, 256);
  cvt_bf_T<<<ge(256 * 64),  b256, 0, stream>>>(Woff, wofft, 256, 64);
  cvt_bf_T<<<ge(256 * 32),  b256, 0, stream>>>(Wattn, wattnt, 256, 32);
  cvt_bf_T<<<ge(512 * 256), b256, 0, stream>>>(W1,   w1t,   256, 512);
  cvt_bf_T<<<ge(512 * 256), b256, 0, stream>>>(W2,   w2t,   512, 256);
  cvt_bf<<<ge(3600 * 256), b256, 0, stream>>>(queries, qbf, 3600 * 256);
  cvt_bf<<<ge(160000 * 256), b256, 0, stream>>>(bev, bevbf, 160000 * 256);

  // self-attention
  add2_bf<<<ge(3600 * 256), b256, 0, stream>>>(queries, og, qkbf, 3600 * 256);
  gemm_bf16<0><<<gg(3600, 256), b128, 0, stream>>>(qkbf, wqt, bq,  nullptr, nullptr, qhbf, 3600, 256, 256);
  gemm_bf16<0><<<gg(3600, 256), b128, 0, stream>>>(qkbf, wkt, bk,  nullptr, nullptr, khbf, 3600, 256, 256);
  gemm_bf16<0><<<gg(3600, 256), b128, 0, stream>>>(qbf,  wvt, bvp, nullptr, nullptr, vhbf, 3600, 256, 256);
  mha_flash<<<dim3(57, 8, 4), b32, 0, stream>>>(qhbf, khbf, vhbf, mask, attnbf);
  gemm_bf16<0><<<gg(3600, 256), b128, 0, stream>>>(attnbf, wot, bo, nullptr, out1, nullptr, 3600, 256, 256);
  add_ln<<<3600, b256, 0, stream>>>(queries, out1, ln1g, ln1b, out2, nullptr);

  // deformable attention
  add2_bf<<<ge(3600 * 256), b256, 0, stream>>>(out2, og, augbf, 3600 * 256);
  gemm_bf16<0><<<gg(160000, 256), b128, 0, stream>>>(bevbf, wvalt, bval, nullptr, nullptr, valbf, 160000, 256, 256);
  gemm_bf16<0><<<gg(3600, 64),  b128, 0, stream>>>(augbf, wofft,  boff,  nullptr, offb, nullptr, 3600, 64, 256);
  gemm_bf16<0><<<gg(3600, 32),  b128, 0, stream>>>(augbf, wattnt, battn, nullptr, awb,  nullptr, 3600, 32, 256);
  deform_sample<<<3600, b256, 0, stream>>>(valbf, offb, awb, ref, sampbf);
  gemm_bf16<2><<<gg(3600, 256), b128, 0, stream>>>(sampbf, wdot, bdo, mask, out3, nullptr, 3600, 256, 256);
  add_ln<<<3600, b256, 0, stream>>>(out2, out3, ln2g, ln2b, out4, o4bf);

  // FFN
  gemm_bf16<1><<<gg(3600, 512), b128, 0, stream>>>(o4bf, w1t, b1, nullptr, nullptr, ffn1bf, 3600, 512, 256);
  gemm_bf16<0><<<gg(3600, 256), b128, 0, stream>>>(ffn1bf, w2t, b2, nullptr, out5, nullptr, 3600, 256, 512);
  add_ln<<<3600, b256, 0, stream>>>(out4, out5, ln3g, ln3b, out, nullptr);
}
